// BreakthroughSNN_86457691668501
// MI455X (gfx1250) — compile-verified, matrix operand order checked by reading, not run on hardware
//
#include <hip/hip_runtime.h>

// ---------------------------------------------------------------------------
// BreakthroughSNN forward for MI455X (gfx1250, wave32, WMMA f16->f32)
//
// Shapes: B=8, S=256, T=16, D=512, DS=64, P=B*S=2048, V=32000
// All matmuls are X @ W^T with row-major W => W rows are the B-operand's
// [N,K] layout directly; spikes are 0/1 so f16 operands are exact.
// A-tiles (16x512) are staged once per block into LDS (async-to-LDS when the
// toolchain exposes the gfx1250 builtins, plain vector stores otherwise).
// ---------------------------------------------------------------------------

#define B_   8
#define S_   256
#define T_   16
#define D_   512
#define DS_  64
#define P_   2048
#define V_   32000
#define DECAY 0.60653065971263342f   // exp(-1/2)
#define ADAPT 0.1f

#define HAVE_ASYNC 0
#if defined(__has_builtin)
#if __has_builtin(__builtin_amdgcn_global_load_async_to_lds_b128) && \
    __has_builtin(__builtin_amdgcn_s_wait_asynccnt)
#undef HAVE_ASYNC
#define HAVE_ASYNC 1
#endif
#endif

typedef _Float16 f16x8 __attribute__((ext_vector_type(8)));
typedef _Float16 v16h  __attribute__((ext_vector_type(16)));
typedef float    v8f   __attribute__((ext_vector_type(8)));
typedef int      i32x4 __attribute__((ext_vector_type(4)));

__device__ __forceinline__ v8f zero8() {
  v8f z;
#pragma unroll
  for (int j = 0; j < 8; ++j) z[j] = 0.f;
  return z;
}

// Fragment loader for 16-bit WMMA operands from a row-major [rows, ld] tile.
// Works for A ([M=16, K] rows) and for B given as [N, K] rows (i.e. X @ W^T
// with row-major W).  Lane (r = lane&15, hf = lane>>4):
//   v[j]   <- tile[r][k0 + hf*8 + j]        (j = 0..7)
//   v[j+8] <- tile[r][k0 + 16 + hf*8 + j]
__device__ __forceinline__ v16h frag_rm(const _Float16* tile, int ld, int k0) {
  const int lane = threadIdx.x & 31;
  const int r    = lane & 15;
  const int hf   = lane >> 4;
  const _Float16* p = tile + (size_t)r * ld + k0 + hf * 8;
  f16x8 lo = *reinterpret_cast<const f16x8*>(p);
  f16x8 hi = *reinterpret_cast<const f16x8*>(p + 16);
  v16h v;
#pragma unroll
  for (int j = 0; j < 8; ++j) { v[j] = lo[j]; v[j + 8] = hi[j]; }
  return v;
}

// B operand stored K-major ([K, N] row-major), K rows limited to `krows`
// (rows >= krows are zero-filled; used for attn(16) x v padded to K=32).
__device__ __forceinline__ v16h frag_km(const _Float16* base, int ldn, int ncol,
                                        int krows) {
  const int lane = threadIdx.x & 31;
  const int n    = lane & 15;
  const int hf   = lane >> 4;
  v16h v;
#pragma unroll
  for (int j = 0; j < 8; ++j) {
    int k0j = hf * 8 + j;
    int k1j = 16 + hf * 8 + j;
    v[j]     = (k0j < krows) ? base[(size_t)k0j * ldn + ncol + n] : (_Float16)0.f;
    v[j + 8] = (k1j < krows) ? base[(size_t)k1j * ldn + ncol + n] : (_Float16)0.f;
  }
  return v;
}

__device__ __forceinline__ v8f wmma16(v16h a, v16h b, v8f c) {
  return __builtin_amdgcn_wmma_f32_16x16x32_f16(false, a, false, b, (short)0, c,
                                                false, false);
}

// Issue copy of a 16-row x 512-col f16 tile (16 KB) from global to LDS.
// Uses gfx1250 GLOBAL_LOAD_ASYNC_TO_LDS_B128 (ASYNCcnt) when available.
__device__ __forceinline__ void stage_issue(_Float16 (*dst)[D_],
                                            const _Float16* src,
                                            size_t row_stride, int nthreads) {
  const int chunks = 16 * (D_ / 8);  // 1024 x 16-byte chunks
  for (int ch = (int)threadIdx.x; ch < chunks; ch += nthreads) {
    int m  = ch >> 6;
    int cc = (ch & 63) * 8;
    const _Float16* g = src + (size_t)m * row_stride + cc;
    _Float16* l = &dst[m][cc];
#if HAVE_ASYNC
    __builtin_amdgcn_global_load_async_to_lds_b128(
        (__attribute__((address_space(1))) i32x4*)(const void*)g,
        (__attribute__((address_space(3))) i32x4*)(void*)l, 0, 0);
#else
    *reinterpret_cast<f16x8*>(l) = *reinterpret_cast<const f16x8*>(g);
#endif
  }
}

__device__ __forceinline__ void stage_wait() {
#if HAVE_ASYNC
  __builtin_amdgcn_s_wait_asynccnt(0);
#endif
}

// ------------------------------- utilities ---------------------------------

__global__ void k_f32_to_f16(const float* __restrict__ s, _Float16* __restrict__ d,
                             long n) {
  long i = (long)blockIdx.x * blockDim.x + threadIdx.x;
  if (i < n) d[i] = (_Float16)s[i];
}
__global__ void k_fill_f32(float* p, float v, long n) {
  long i = (long)blockIdx.x * blockDim.x + threadIdx.x;
  if (i < n) p[i] = v;
}
__global__ void k_fill_f16(_Float16* p, float v, long n) {
  long i = (long)blockIdx.x * blockDim.x + threadIdx.x;
  if (i < n) p[i] = (_Float16)v;
}
__global__ void k_fill_i32(int* p, int v, long n) {
  long i = (long)blockIdx.x * blockDim.x + threadIdx.x;
  if (i < n) p[i] = v;
}

// ------------------------- embed + TTFS encode -----------------------------
// x0[b][t][s][d] = (t == round(10*(1-sigmoid(scale*emb[id][d])))) ? 1 : 0
__global__ void k_embed_ttfs(const int* __restrict__ ids,
                             const float* __restrict__ emb,
                             const float* __restrict__ scaling,
                             _Float16* __restrict__ x0, int* __restrict__ flags0) {
  __shared__ int sf[16];
  if (threadIdx.x < 16) sf[threadIdx.x] = 0;
  __syncthreads();
  long gid = (long)blockIdx.x * blockDim.x + threadIdx.x;  // B*S*D threads
  int d  = (int)(gid & 511);
  long p = gid >> 9;
  int b = (int)(p >> 8), s = (int)(p & 255);
  float e   = emb[(long)ids[p] * D_ + d];
  float act = 1.f / (1.f + __expf(-scaling[0] * e));
  int st = (int)rintf(10.f * (1.f - act));     // ties-to-even like jnp.round
  st = st < 0 ? 0 : (st > T_ - 1 ? T_ - 1 : st);
  size_t base = (((size_t)b * T_) * S_ + s) * D_ + d;
  const size_t tstr = (size_t)S_ * D_;
#pragma unroll
  for (int t = 0; t < T_; ++t)
    x0[base + (size_t)t * tstr] = (t == st) ? (_Float16)1.f : (_Float16)0.f;
  atomicOr(&sf[st], 1);
  __syncthreads();
  if (threadIdx.x < 16 && sf[threadIdx.x]) atomicOr(&flags0[threadIdx.x], 1);
}

// ------------------------------ SSM step -----------------------------------
// One block handles one 16-row tile of P_=2048 (b, s0..s0+15), 4 waves.
// Stage 1: su = h @ A^T (+ x_t @ B^T if active); LIF on state; h <- spikes.
// Stage 2 (active): ou = spikes @ C^T + x_t @ D^T; LIF on output; write x_out.
//         (silent): x_out tile = 0; ov/thr_o untouched.
// The x_t tile is staged into LDS once (async DMA overlapped with the h@A^T
// WMMA work) and reused by both stages across all 4 waves.
__global__ void __launch_bounds__(128) k_ssm_step(
    const _Float16* __restrict__ xin, _Float16* __restrict__ xout,
    _Float16* __restrict__ h, float* __restrict__ sv, float* __restrict__ ov,
    const float* __restrict__ thr_s, const float* __restrict__ thr_o,
    float* __restrict__ acc_s, float* __restrict__ acc_o,
    const int* __restrict__ flagIn, int* __restrict__ flagOut, int t_,
    const _Float16* __restrict__ A16, const _Float16* __restrict__ B16,
    const _Float16* __restrict__ C16, const _Float16* __restrict__ D16) {
  __shared__ _Float16 sspike[16][DS_];
  __shared__ _Float16 xt[16][D_];
  const int p0 = blockIdx.x * 16;
  const int b = p0 >> 8, s0 = p0 & 255;
  const size_t xbase = ((size_t)(b * T_ + t_) * S_ + s0) * D_;
  const int w    = threadIdx.x >> 5;
  const int lane = threadIdx.x & 31;
  const int nloc = lane & 15;
  const int mb   = (lane >> 4) * 8;
  const bool active = (flagIn[t_] != 0);   // uniform across block

  if (active) stage_issue(xt, xin + xbase, D_, 128);  // overlap with h@A^T

  // ---- stage 1 ----
  v8f acc1 = zero8();
  {
    const _Float16* htile = h + (size_t)p0 * DS_;
#pragma unroll
    for (int k0 = 0; k0 < DS_; k0 += 32) {
      v16h a  = frag_rm(htile, DS_, k0);
      v16h bb = frag_rm(A16 + (size_t)(w * 16) * DS_, DS_, k0);
      acc1 = wmma16(a, bb, acc1);
    }
  }
  stage_wait();
  __syncthreads();  // h reads done before h writes; xt visible to all waves
  if (active) {
    for (int k0 = 0; k0 < D_; k0 += 32) {
      v16h a  = frag_rm(&xt[0][0], D_, k0);
      v16h bb = frag_rm(B16 + (size_t)(w * 16) * D_, D_, k0);
      acc1 = wmma16(a, bb, acc1);
    }
  }
  {
    const int col = w * 16 + nloc;
    const float thr = thr_s[col];
    float cnt = 0.f;
#pragma unroll
    for (int r = 0; r < 8; ++r) {
      const size_t idx = (size_t)(p0 + mb + r) * DS_ + col;
      float vpot = sv[idx] * DECAY + acc1[r];
      float sp   = (vpot - thr >= 0.f) ? 1.f : 0.f;
      sv[idx] = vpot * (1.f - sp);
      _Float16 sp16 = (_Float16)sp;
      h[idx] = sp16;
      sspike[mb + r][col] = sp16;
      cnt += sp;
    }
    atomicAdd(&acc_s[col], cnt);
  }
  __syncthreads();

  // ---- stage 2 ----
  if (active) {
    v8f acc2[8];
#pragma unroll
    for (int i = 0; i < 8; ++i) acc2[i] = zero8();
#pragma unroll
    for (int k0 = 0; k0 < DS_; k0 += 32) {           // spikes @ C^T (K=64)
      v16h a = frag_rm(&sspike[0][0], DS_, k0);
#pragma unroll
      for (int i = 0; i < 8; ++i) {
        v16h bb = frag_rm(C16 + (size_t)(w * 128 + i * 16) * DS_, DS_, k0);
        acc2[i] = wmma16(a, bb, acc2[i]);
      }
    }
    for (int k0 = 0; k0 < D_; k0 += 32) {            // x_t @ D^T (K=512)
      v16h a = frag_rm(&xt[0][0], D_, k0);
#pragma unroll
      for (int i = 0; i < 8; ++i) {
        v16h bb = frag_rm(D16 + (size_t)(w * 128 + i * 16) * D_, D_, k0);
        acc2[i] = wmma16(a, bb, acc2[i]);
      }
    }
    float any = 0.f;
#pragma unroll
    for (int i = 0; i < 8; ++i) {
      const int col = w * 128 + i * 16 + nloc;
      const float thr = thr_o[col];
      float cnt = 0.f;
#pragma unroll
      for (int r = 0; r < 8; ++r) {
        const int m = mb + r;
        const size_t oidx = (size_t)(p0 + m) * D_ + col;
        float vpot = ov[oidx] * DECAY + acc2[i][r];
        float sp   = (vpot - thr >= 0.f) ? 1.f : 0.f;
        ov[oidx] = vpot * (1.f - sp);
        xout[xbase + (size_t)m * D_ + col] = (_Float16)sp;
        cnt += sp;
      }
      atomicAdd(&acc_o[col], cnt);
      any += cnt;
    }
    if (flagOut != nullptr && any > 0.f) atomicOr(&flagOut[t_], 1);
  } else {
    const f16x8 z = {};
    for (int ch = threadIdx.x; ch < 16 * (D_ / 8); ch += 128) {
      int m  = ch >> 6;          // D_/8 = 64 chunks per row
      int cc = (ch & 63) * 8;
      *reinterpret_cast<f16x8*>(xout + xbase + (size_t)m * D_ + cc) = z;
    }
  }
}

// threshold adaptation (once per (layer, t)); silent steps leave thr_o alone
__global__ void k_thr_update(float* __restrict__ thr_s, float* __restrict__ thr_o,
                             float* __restrict__ acc_s, float* __restrict__ acc_o,
                             const int* __restrict__ flag) {
  const int tid = threadIdx.x;          // 512 threads
  const int f = *flag;
  if (tid < DS_) {
    thr_s[tid] += ADAPT * (acc_s[tid] * (1.f / (float)P_) - 0.1f);
    acc_s[tid] = 0.f;
  }
  if (f) thr_o[tid] += ADAPT * (acc_o[tid] * (1.f / (float)P_) - 0.1f);
  acc_o[tid] = 0.f;
}

// ------------------- fused temporal attention + mean -----------------------
// One block per (b,s) position, 8 waves = 8 heads.  T=16 rows == one WMMA M.
__global__ void __launch_bounds__(256) k_attention(
    const _Float16* __restrict__ x2, const _Float16* __restrict__ Wq16,
    const _Float16* __restrict__ Wk16, const _Float16* __restrict__ Wv16,
    const _Float16* __restrict__ Wo16, const float* __restrict__ bq,
    const float* __restrict__ bk, const float* __restrict__ bv,
    const float* __restrict__ bo, _Float16* __restrict__ ti) {
  __shared__ _Float16 xr[T_][D_];
  __shared__ _Float16 qs[T_][D_];        // reused as av after scores
  __shared__ _Float16 ks[T_][D_];
  __shared__ _Float16 vs[T_][D_];
  __shared__ _Float16 at[8][16][32];     // per-head attn, K padded to 32
  const int p = blockIdx.x;
  const int b = p >> 8, s = p & 255;
  const size_t xbase = (((size_t)b * T_) * S_ + s) * D_;
  const size_t tstr  = (size_t)S_ * D_;
  const int w    = threadIdx.x >> 5;
  const int lane = threadIdx.x & 31;
  const int nloc = lane & 15;
  const int mb   = (lane >> 4) * 8;

  stage_issue(xr, x2 + xbase, tstr, 256);   // async copy x rows (t-strided)
  stage_wait();
  __syncthreads();

  // q/k/v projections: 96 16x16 tiles, 12 per wave
  for (int j = 0; j < 12; ++j) {
    int tile = w + 8 * j;
    int mat  = tile >> 5;                 // 0=q, 1=k, 2=v
    int ncol = (tile & 31) * 16;
    const _Float16* W = (mat == 0) ? Wq16 : (mat == 1) ? Wk16 : Wv16;
    const float* bias = (mat == 0) ? bq : (mat == 1) ? bk : bv;
    _Float16(*dst)[D_] = (mat == 0) ? qs : (mat == 1) ? ks : vs;
    v8f acc = zero8();
    for (int k0 = 0; k0 < D_; k0 += 32) {
      v16h a  = frag_rm(&xr[0][0], D_, k0);
      v16h bb = frag_rm(W + (size_t)ncol * D_, D_, k0);
      acc = wmma16(a, bb, acc);
    }
    int col = ncol + nloc;
    float bval = bias[col];
#pragma unroll
    for (int r = 0; r < 8; ++r) dst[mb + r][col] = (_Float16)(acc[r] + bval);
  }
  __syncthreads();

  // scores for head w: q_h [16,64] x k_h^T -> [16,16], then row softmax
  v8f sc = zero8();
#pragma unroll
  for (int k0 = 0; k0 < 64; k0 += 32) {
    v16h a  = frag_rm(&qs[0][w * 64], D_, k0);
    v16h bb = frag_rm(&ks[0][w * 64], D_, k0);
    sc = wmma16(a, bb, sc);
  }
#pragma unroll
  for (int r = 0; r < 8; ++r) {
    float v = sc[r] * 0.125f;            // 1/sqrt(64)
    float mx = v;
#pragma unroll
    for (int off = 1; off < 16; off <<= 1) mx = fmaxf(mx, __shfl_xor(mx, off, 32));
    float e = __expf(v - mx);
    float sum = e;
#pragma unroll
    for (int off = 1; off < 16; off <<= 1) sum += __shfl_xor(sum, off, 32);
    at[w][mb + r][nloc]      = (_Float16)(e / sum);
    at[w][mb + r][16 + nloc] = (_Float16)0.f;       // zero pad K=16..31
  }
  // av = attn x v_h -> into qs columns of this head (wave-local)
  {
    v16h a = frag_rm(&at[w][0][0], 32, 0);
#pragma unroll
    for (int i = 0; i < 4; ++i) {
      int ncol = w * 64 + i * 16;
      v16h bb = frag_km(&vs[0][0], D_, ncol, 16);
      v8f acc = wmma16(a, bb, zero8());
      int col = ncol + nloc;
#pragma unroll
      for (int r = 0; r < 8; ++r) qs[mb + r][col] = (_Float16)acc[r];
    }
  }
  __syncthreads();

  // out-projection + residual + mean over T
  for (int i = 0; i < 4; ++i) {
    int ncol = (w * 4 + i) * 16;
    v8f acc = zero8();
    for (int k0 = 0; k0 < D_; k0 += 32) {
      v16h a  = frag_rm(&qs[0][0], D_, k0);          // av
      v16h bb = frag_rm(Wo16 + (size_t)ncol * D_, D_, k0);
      acc = wmma16(a, bb, acc);
    }
    int col = ncol + nloc;
    float s8 = 0.f;
#pragma unroll
    for (int r = 0; r < 8; ++r) s8 += acc[r] + (float)xr[mb + r][col];
    float tot = s8 + __shfl_xor(s8, 16, 32);         // both row halves
    ti[(size_t)p * D_ + col] = (_Float16)(tot * (1.f / 16.f) + bo[col]);
  }
}

// ------------------------------ vocab GEMM ---------------------------------
// logits[2048, 32000] = ti[2048,512] x Wout^T + bout.  4 waves/block,
// each wave: 16 rows x 64 cols; the shared A tile lives in LDS.
__global__ void __launch_bounds__(128) k_logits(
    const _Float16* __restrict__ ti, const _Float16* __restrict__ Wout16,
    const float* __restrict__ bout, float* __restrict__ out) {
  __shared__ _Float16 atile[16][D_];
  const int p0   = blockIdx.y * 16;
  const int w    = threadIdx.x >> 5;
  const int lane = threadIdx.x & 31;
  const int nloc = lane & 15;
  const int mb   = (lane >> 4) * 8;
  const int nbase = blockIdx.x * 256 + w * 64;

  stage_issue(atile, ti + (size_t)p0 * D_, D_, 128);
  stage_wait();
  __syncthreads();

  v8f acc[4];
#pragma unroll
  for (int i = 0; i < 4; ++i) acc[i] = zero8();
  for (int k0 = 0; k0 < D_; k0 += 32) {
    v16h a = frag_rm(&atile[0][0], D_, k0);
    __builtin_prefetch(Wout16 + (size_t)nbase * D_ + k0 + 256, 0, 1);
#pragma unroll
    for (int i = 0; i < 4; ++i) {
      v16h bb = frag_rm(Wout16 + (size_t)(nbase + i * 16) * D_, D_, k0);
      acc[i] = wmma16(a, bb, acc[i]);
    }
  }
#pragma unroll
  for (int i = 0; i < 4; ++i) {
    int col = nbase + i * 16 + nloc;
    float bv = bout[col];
#pragma unroll
    for (int r = 0; r < 8; ++r)
      out[(size_t)(p0 + mb + r) * V_ + col] = acc[i][r] + bv;
  }
}

// ------------------------------- launcher ----------------------------------

extern "C" void kernel_launch(void* const* d_in, const int* in_sizes, int n_in,
                              void* d_out, int out_size, void* d_ws, size_t ws_size,
                              hipStream_t stream) {
  (void)in_sizes; (void)n_in; (void)out_size; (void)ws_size;
  const int*   ids     = (const int*)d_in[0];
  const float* emb     = (const float*)d_in[1];
  const float* scaling = (const float*)d_in[2];
  const float* As      = (const float*)d_in[3];
  const float* Bs      = (const float*)d_in[4];
  const float* Cs      = (const float*)d_in[5];
  const float* Ds      = (const float*)d_in[6];
  const float* Wq      = (const float*)d_in[7];
  const float* bq      = (const float*)d_in[8];
  const float* Wk      = (const float*)d_in[9];
  const float* bk      = (const float*)d_in[10];
  const float* Wv      = (const float*)d_in[11];
  const float* bv      = (const float*)d_in[12];
  const float* Wo      = (const float*)d_in[13];
  const float* bo      = (const float*)d_in[14];
  const float* Wout    = (const float*)d_in[15];
  const float* bout    = (const float*)d_in[16];
  float* out = (float*)d_out;

  char* ws = (char*)d_ws;
  size_t off = 0;
  auto take = [&](size_t bytes) -> char* {
    char* r = ws + off;
    off = (off + bytes + 255) & ~(size_t)255;
    return r;
  };
  const size_t XN = (size_t)B_ * T_ * S_ * D_;   // 16,777,216
  _Float16* xA     = (_Float16*)take(XN * 2);
  _Float16* xB     = (_Float16*)take(XN * 2);
  _Float16* Wout16 = (_Float16*)take((size_t)V_ * D_ * 2);
  _Float16* Wq16   = (_Float16*)take((size_t)D_ * D_ * 2);
  _Float16* Wk16   = (_Float16*)take((size_t)D_ * D_ * 2);
  _Float16* Wv16   = (_Float16*)take((size_t)D_ * D_ * 2);
  _Float16* Wo16   = (_Float16*)take((size_t)D_ * D_ * 2);
  _Float16* Ds16   = (_Float16*)take((size_t)2 * D_ * D_ * 2);
  _Float16* Cs16   = (_Float16*)take((size_t)2 * D_ * DS_ * 2);
  _Float16* Bs16   = (_Float16*)take((size_t)2 * DS_ * D_ * 2);
  _Float16* As16   = (_Float16*)take((size_t)2 * DS_ * DS_ * 2);
  _Float16* h16    = (_Float16*)take((size_t)P_ * DS_ * 2);
  float*    sv     = (float*)take((size_t)P_ * DS_ * 4);
  float*    ov     = (float*)take((size_t)P_ * D_ * 4);
  float*    thr_s  = (float*)take((size_t)2 * DS_ * 4);
  float*    thr_o  = (float*)take((size_t)2 * D_ * 4);
  float*    acc_s  = (float*)take((size_t)DS_ * 4);
  float*    acc_o  = (float*)take((size_t)D_ * 4);
  int*      flags  = (int*)take((size_t)2 * T_ * 4);
  _Float16* ti16   = (_Float16*)take((size_t)P_ * D_ * 2);

  auto cvt = [&](const float* s, _Float16* d, long n) {
    k_f32_to_f16<<<dim3((unsigned)((n + 255) / 256)), 256, 0, stream>>>(s, d, n);
  };
  cvt(Wout, Wout16, (long)V_ * D_);
  cvt(Wq, Wq16, (long)D_ * D_);
  cvt(Wk, Wk16, (long)D_ * D_);
  cvt(Wv, Wv16, (long)D_ * D_);
  cvt(Wo, Wo16, (long)D_ * D_);
  cvt(Ds, Ds16, (long)2 * D_ * D_);
  cvt(Cs, Cs16, (long)2 * D_ * DS_);
  cvt(Bs, Bs16, (long)2 * DS_ * D_);
  cvt(As, As16, (long)2 * DS_ * DS_);

  k_fill_f32<<<1, 128, 0, stream>>>(thr_s, 1.f, 2 * DS_);
  k_fill_f32<<<4, 256, 0, stream>>>(thr_o, 1.f, 2 * D_);
  k_fill_f32<<<1, 64, 0, stream>>>(acc_s, 0.f, DS_);
  k_fill_f32<<<2, 256, 0, stream>>>(acc_o, 0.f, D_);
  k_fill_i32<<<1, 32, 0, stream>>>(flags, 0, 2 * T_);

  // embed + TTFS encode (writes xA, layer-0 active flags)
  k_embed_ttfs<<<dim3(4096), 256, 0, stream>>>(ids, emb, scaling, xA, flags);

  // two SSM layers, sequential over T with threshold adaptation
  for (int layer = 0; layer < 2; ++layer) {
    const _Float16* xin = (layer == 0) ? xA : xB;
    _Float16* xout      = (layer == 0) ? xB : xA;
    const int* fin = flags + layer * T_;
    int* fout = (layer == 0) ? (flags + T_) : nullptr;
    // fresh carry per layer (reference re-zeroes buffers every forward)
    k_fill_f16<<<dim3((P_ * DS_ + 255) / 256), 256, 0, stream>>>(h16, 0.f, (long)P_ * DS_);
    k_fill_f32<<<dim3((P_ * DS_ + 255) / 256), 256, 0, stream>>>(sv, 0.f, (long)P_ * DS_);
    k_fill_f32<<<dim3((P_ * D_ + 255) / 256), 256, 0, stream>>>(ov, 0.f, (long)P_ * D_);
    for (int t = 0; t < T_; ++t) {
      k_ssm_step<<<dim3(P_ / 16), 128, 0, stream>>>(
          xin, xout, h16, sv, ov, thr_s + layer * DS_, thr_o + layer * D_,
          acc_s, acc_o, fin, fout, t,
          As16 + (size_t)layer * DS_ * DS_, Bs16 + (size_t)layer * DS_ * D_,
          Cs16 + (size_t)layer * D_ * DS_, Ds16 + (size_t)layer * D_ * D_);
      k_thr_update<<<1, 512, 0, stream>>>(thr_s + layer * DS_, thr_o + layer * D_,
                                          acc_s, acc_o, fin + t);
    }
  }

  // temporal attention + residual + time-mean (input is xA after layer 1)
  k_attention<<<dim3(P_), 256, 0, stream>>>(xA, Wq16, Wk16, Wv16, Wo16, bq, bk,
                                            bv, bo, ti16);

  // vocab projection
  k_logits<<<dim3(V_ / 256, P_ / 16), 128, 0, stream>>>(ti16, Wout16, bout, out);
}
